// EdgeDecoder_72164040508151
// MI455X (gfx1250) — matrix operand order repaired and checked by reading, not compiled, not run on hardware
//
#include <hip/hip_runtime.h>
#include <hip/hip_bf16.h>

// ---------------------------------------------------------------------------
// EdgeDecoder for MI455X (gfx1250, wave32, WMMA, async global->LDS)
//
//   z   = x[src] @ w1a^T + x[dst] @ w1b^T   (+ b1, which cancels through BN)
//   z   = BN(z) * gamma + beta ; relu ; out = z @ w2^T + b2
//
// Strategy:
//   * x converted once to bf16 (25.6 MB -> L2-resident on 192MB L2).
//   * w1^T pre-packed into exact per-lane WMMA-B fragment layout (64 KB).
//   * Per block (128 edges x 128 feats, 8 waves): gathered src|dst rows are
//     staged ONCE into a 64KB LDS tile via GLOBAL_LOAD_ASYNC_TO_LDS_B128
//     (ASYNCcnt), killing the 8x per-wave L2 gather redundancy. WMMA A
//     fragments then come from LDS (ds_load_b128), B from L2.
//   * Pass 1: GEMM -> per-block per-feature sum/sumsq partials (no atomics),
//     deterministic tree reduction, fold BN+gamma/beta into scale/shift.
//   * Pass 2: recompute GEMM (cheaper than spilling 307MB of z), apply
//     scale/shift, relu, *w2, shuffle-reduce across features, +b2 -> out.
// ---------------------------------------------------------------------------

typedef __attribute__((ext_vector_type(8)))  __bf16 v8bf;
typedef __attribute__((ext_vector_type(16))) __bf16 v16bf;
typedef __attribute__((ext_vector_type(8)))  float  v8f;

#define NN        100000          // nodes
#define KH        128             // hidden dim
#define NE        600000          // edges
#define BN_EPS    1e-5f
#define NBLK      ((NE + 127) / 128)   // 4688 edge tiles
#define RED1      64                   // stage-1 reduction blocks

// ---- workspace layout (bytes) ----
#define XBF_OFF    0
#define XBF_BYTES  (NN * KH * 2)                    // 25,600,000
#define BFRAG_OFF  (XBF_OFF + XBF_BYTES)            // 256B aligned
#define BFRAG_BYTES (8 * 8 * 32 * 16 * 2)           // 65,536
#define STATS_OFF  (BFRAG_OFF + BFRAG_BYTES)        // 512 floats: sum|sq|scale|shift
#define STATS_BYTES (512 * 4)
#define PART_OFF   (STATS_OFF + STATS_BYTES)        // NBLK x 256 floats
#define PART_BYTES (NBLK * 256 * 4)
#define PART2_OFF  (PART_OFF + PART_BYTES)          // RED1 x 256 floats

static __device__ __forceinline__ unsigned short f2bf(float f) {
    unsigned int u = __float_as_uint(f);
    u += 0x7FFFu + ((u >> 16) & 1u);        // round to nearest even
    return (unsigned short)(u >> 16);
}

// ---------------------------------------------------------------------------
// Prep: x fp32 -> bf16
__global__ __launch_bounds__(256) void k_convert_x(const float* __restrict__ x,
                                                   unsigned short* __restrict__ xbf) {
    int i = blockIdx.x * 256 + threadIdx.x;
    if (i < NN * KH) xbf[i] = f2bf(x[i]);
}

// Prep: pack B = w1^T (256x128) into per-lane WMMA-B fragments.
// B fragment layout (16-bit B, 32x16, wave32): VGPR v, lane L holds packed
// {K = kb*32 + (L>=16?8:0) + 2(v&3) + (v>=4?16:0), +1} for column N = L&15.  Stored per lane
// as 16 contiguous halves: bfrag[((kb*8+c)*32+L)*16 + h] = w1[n, k],
// n = c*16 + (L&15), k = kb*32 + ((L&16)?8:0) + (h&7) + ((h&8)?16:0).
__global__ __launch_bounds__(256) void k_build_b(const float* __restrict__ w1,
                                                 unsigned short* __restrict__ bfrag,
                                                 float* __restrict__ stats) {
    int t = blockIdx.x * 256 + threadIdx.x;          // [0, 32768)
    if (blockIdx.x == 0 && threadIdx.x < 256) stats[threadIdx.x] = 0.0f;
    int h    = t & 15;
    int lane = (t >> 4) & 31;
    int c    = (t >> 9) & 7;
    int kb   = t >> 12;                              // 0..7
    int k = kb * 32 + ((lane & 16) ? 8 : 0) + (h & 7) + ((h & 8) ? 16 : 0);
    int n = c * 16 + (lane & 15);
    bfrag[t] = f2bf(w1[n * 256 + k]);
}

// ---------------------------------------------------------------------------
// Stage 128 gathered edge rows (src|dst, 512B/edge = 64KB) into LDS via
// CDNA5 async global->LDS loads (ASYNCcnt path, no VGPR staging).
__device__ __forceinline__ void stage_tile(const unsigned short* __restrict__ xbf,
                                           const long long* __restrict__ eidx,
                                           int e0, int tid,
                                           unsigned short* sA /*shared*/,
                                           int* lrow /*shared [128][2]*/) {
    if (tid < 128) {
        int e = e0 + tid;
        if (e >= NE) e = NE - 1;                     // clamp; masked later
        lrow[2 * tid]     = (int)eidx[2 * e];
        lrow[2 * tid + 1] = (int)eidx[2 * e + 1];
    }
    __syncthreads();
    // LDS byte offset: ISA maps generic-shared addr[31:0] -> LDS address.
    unsigned abase = (unsigned)(size_t)(void*)sA;
    int cch  = tid & 31;                             // 32 x 16B chunks per edge
    int wv   = tid >> 5;
#pragma unroll
    for (int i = 0; i < 16; ++i) {
        int j   = i * 8 + wv;                        // edge slot 0..127
        int row = lrow[2 * j + (cch >> 4)];          // src (0..15) | dst (16..31)
        const unsigned short* gp = xbf + row * KH + (cch & 15) * 8;
        unsigned ldsoff = abase + (unsigned)(j * 512 + cch * 16);
        asm volatile("global_load_async_to_lds_b128 %0, %1, off"
                     :: "v"(ldsoff), "v"(gp) : "memory");
    }
    asm volatile("s_wait_asynccnt 0x0" ::: "memory");
    __syncthreads();
}

// ---------------------------------------------------------------------------
// Core: one wave computes a 128-edge x 16-feature slab as 8 WMMA row tiles,
// A fragments from the staged LDS tile.  A layout (16-bit A, 16x32): lane L
// = row (L&15); lanes 0-15 take K {0..7,16..23}, lanes 16-31 {8..15,24..31}.
__device__ __forceinline__ void gemm_tile(const unsigned short* sA /*shared*/,
                                          const unsigned short* __restrict__ bfrag,
                                          int lane, int c, v8f acc[8]) {
#pragma unroll
    for (int r = 0; r < 8; ++r) acc[r] = (v8f){0.f,0.f,0.f,0.f,0.f,0.f,0.f,0.f};
    const int hi = (lane & 16) ? 8 : 0;              // per-lane K sub-offset
    const int jl = lane & 15;                        // row within tile
#pragma unroll
    for (int kb = 0; kb < 8; ++kb) {
        const v8bf* bp = (const v8bf*)(bfrag + (((kb * 8 + c) * 32 + lane) << 4));
        v16bf b = __builtin_shufflevector(bp[0], bp[1],
                    0,1,2,3,4,5,6,7,8,9,10,11,12,13,14,15);
        int koff = ((kb & 4) << 5) + (kb & 3) * 32 + hi;  // src half | dst half
#pragma unroll
        for (int r = 0; r < 8; ++r) {
            const v8bf* ap = (const v8bf*)(sA + (r * 16 + jl) * 256 + koff);
            v16bf a = __builtin_shufflevector(ap[0], ap[2],   // +0 and +16 halves
                        0,1,2,3,4,5,6,7,8,9,10,11,12,13,14,15);
            acc[r] = __builtin_amdgcn_wmma_f32_16x16x32_bf16(
                         false, a, false, b, (short)0, acc[r], false, false);
        }
    }
}

// ---------------------------------------------------------------------------
// Pass 1: GEMM + per-block per-feature sum / sumsq partials (deterministic).
__global__ __launch_bounds__(256) void k_gemm_stats(const unsigned short* __restrict__ xbf,
                                                    const unsigned short* __restrict__ bfrag,
                                                    const long long* __restrict__ eidx,
                                                    float* __restrict__ part) {
    __shared__ unsigned short sA[128 * 256];          // 64KB staged tile
    __shared__ int   lrow[128 * 2];
    __shared__ float lsum[128], lsq[128];
    const int tid  = threadIdx.x;
    const int lane = tid & 31, c = tid >> 5;
    const int e0   = blockIdx.x * 128;

    stage_tile(xbf, eidx, e0, tid, sA, lrow);

    v8f acc[8];
    gemm_tile(sA, bfrag, lane, c, acc);

    const int mbase = (lane & 16) ? 8 : 0;           // C layout: M = i + mbase
    float s = 0.f, s2 = 0.f;
#pragma unroll
    for (int r = 0; r < 8; ++r)
#pragma unroll
        for (int i = 0; i < 8; ++i) {
            float v = acc[r][i];
            int e = e0 + r * 16 + mbase + i;
            v = (e < NE) ? v : 0.f;
            s += v; s2 += v * v;
        }
    s  += __shfl_xor(s, 16, 32);                     // fold the two M-halves
    s2 += __shfl_xor(s2, 16, 32);
    if (lane < 16) {                                  // feature c*16+lane: unique owner
        lsum[c * 16 + lane] = s;
        lsq [c * 16 + lane] = s2;
    }
    __syncthreads();
    if (tid < 128) {
        part[blockIdx.x * 256 + tid]       = lsum[tid];
        part[blockIdx.x * 256 + 128 + tid] = lsq[tid];
    }
}

// Deterministic two-stage tree reduction of the NBLK x 256 partials.
__global__ __launch_bounds__(256) void k_reduce1(const float* __restrict__ part,
                                                 float* __restrict__ part2) {
    int t = threadIdx.x;
    float s = 0.f;
    for (int p = blockIdx.x; p < NBLK; p += RED1) s += part[p * 256 + t];
    part2[blockIdx.x * 256 + t] = s;
}
__global__ __launch_bounds__(256) void k_reduce2(const float* __restrict__ part2,
                                                 float* __restrict__ stats) {
    int t = threadIdx.x;
    float s = 0.f;
    for (int b = 0; b < RED1; ++b) s += part2[b * 256 + t];
    stats[t] = s;
}

// Fold BN + gamma/beta into per-feature scale/shift.
__global__ void k_finalize(float* __restrict__ stats,
                           const float* __restrict__ gamma,
                           const float* __restrict__ beta) {
    int t = threadIdx.x;                              // 128
    float inv  = 1.0f / (float)NE;
    float mean = stats[t] * inv;
    float var  = fmaxf(stats[128 + t] * inv - mean * mean, 0.0f);
    float sc   = gamma[t] * rsqrtf(var + BN_EPS);
    stats[256 + t] = sc;
    stats[384 + t] = beta[t] - mean * sc;
}

// ---------------------------------------------------------------------------
// Pass 2: recompute GEMM, BN+relu, dot with w2, per-edge output.
__global__ __launch_bounds__(256) void k_gemm_out(const unsigned short* __restrict__ xbf,
                                                  const unsigned short* __restrict__ bfrag,
                                                  const long long* __restrict__ eidx,
                                                  const float* __restrict__ stats,
                                                  const float* __restrict__ w2,
                                                  const float* __restrict__ b2,
                                                  float* __restrict__ out) {
    __shared__ unsigned short sA[128 * 256];          // 64KB staged tile
    __shared__ int   lrow[128 * 2];
    __shared__ float wacc[8][128];                    // per-wave partials
    const int tid  = threadIdx.x;
    const int lane = tid & 31, c = tid >> 5;
    const int n    = c * 16 + (lane & 15);
    const float sc = stats[256 + n];
    const float sh = stats[384 + n];
    const float wn = w2[n];
    const int e0   = blockIdx.x * 128;

    stage_tile(xbf, eidx, e0, tid, sA, lrow);

    v8f acc[8];
    gemm_tile(sA, bfrag, lane, c, acc);

    const int mbase = (lane & 16) ? 8 : 0;
#pragma unroll
    for (int r = 0; r < 8; ++r)
#pragma unroll
        for (int i = 0; i < 8; ++i) {
            float t = fmaf(acc[r][i], sc, sh);
            t = fmaxf(t, 0.0f) * wn;
            t += __shfl_xor(t, 1, 32);                // reduce 16 features
            t += __shfl_xor(t, 2, 32);
            t += __shfl_xor(t, 4, 32);
            t += __shfl_xor(t, 8, 32);
            if ((lane & 15) == 0)
                wacc[c][r * 16 + mbase + i] = t;      // unique slot per wave
        }
    __syncthreads();
    if (tid < 128) {
        float s = b2[0];
#pragma unroll
        for (int w = 0; w < 8; ++w) s += wacc[w][tid]; // fixed order: deterministic
        int e = e0 + tid;
        if (e < NE) out[e] = s;
    }
}

// ---------------------------------------------------------------------------
extern "C" void kernel_launch(void* const* d_in, const int* in_sizes, int n_in,
                              void* d_out, int out_size, void* d_ws, size_t ws_size,
                              hipStream_t stream) {
    const float*     x     = (const float*)d_in[0];
    const long long* eidx  = (const long long*)d_in[1];   // int64 (E,2)
    const float*     w1    = (const float*)d_in[2];
    // d_in[3] = b1: cancels exactly through BatchNorm -> intentionally unused.
    const float*     gamma = (const float*)d_in[4];
    const float*     beta  = (const float*)d_in[5];
    const float*     w2    = (const float*)d_in[6];
    const float*     b2    = (const float*)d_in[7];
    float*           out   = (float*)d_out;

    char* ws = (char*)d_ws;
    unsigned short* xbf   = (unsigned short*)(ws + XBF_OFF);
    unsigned short* bfrag = (unsigned short*)(ws + BFRAG_OFF);
    float*          stats = (float*)(ws + STATS_OFF);
    float*          part  = (float*)(ws + PART_OFF);
    float*          part2 = (float*)(ws + PART2_OFF);

    k_convert_x<<<(NN * KH + 255) / 256, 256, 0, stream>>>(x, xbf);
    k_build_b<<<128, 256, 0, stream>>>(w1, bfrag, stats);
    k_gemm_stats<<<NBLK, 256, 0, stream>>>(xbf, bfrag, eidx, part);
    k_reduce1<<<RED1, 256, 0, stream>>>(part, part2);
    k_reduce2<<<1, 256, 0, stream>>>(part2, stats);
    k_finalize<<<1, 128, 0, stream>>>(stats, gamma, beta);
    k_gemm_out<<<NBLK, 256, 0, stream>>>(xbf, bfrag, eidx, stats, w2, b2, out);
}
